// VGG_Cifar10_16595753632454
// MI455X (gfx1250) — 8-bit matrix operand order repaired and checked by reading, compiled, not run on hardware
//
#include <hip/hip_runtime.h>
#include <cstdint>
#include <cstddef>

typedef __attribute__((ext_vector_type(8))) int v8i;

__device__ __forceinline__ v8i packAu(int2 a01, int2 a23, int2 a45, int2 a67) {
  v8i a;
  a[0] = a01.x; a[1] = a01.y; a[2] = a23.x; a[3] = a23.y;
  a[4] = a45.x; a[5] = a45.y; a[6] = a67.x; a[7] = a67.y;
  return a;
}
__device__ __forceinline__ v8i packB(int4 lo, int4 hi) {
  v8i b;
  b[0] = lo.x; b[1] = lo.y; b[2] = lo.z; b[3] = lo.w;
  b[4] = hi.x; b[5] = hi.y; b[6] = hi.z; b[7] = hi.w;
  return b;
}
__device__ __forceinline__ int8_t sgn8(float v) {
  return (v > 0.f) ? (int8_t)1 : ((v < 0.f) ? (int8_t)-1 : (int8_t)0);
}

__global__ void bconv_gemm_iu8(const int8_t* __restrict__ act,
                               const int8_t* __restrict__ wgt,
                               int* __restrict__ acc,
                               int Bn, int H, int W, int Ci, int Co) {
  const int K = 9 * Ci;
  const int Hp = H + 2, Wp = W + 2;
  const int tilesN = Co >> 5;
  const int nTiles = ((Bn * H * W) >> 4) * tilesN;
  int wave = (int)((blockIdx.x * blockDim.x + threadIdx.x) >> 5);
  int lane = threadIdx.x & 31;
  if (wave >= nTiles) return;
  int tm = (wave / tilesN) << 4;
  int tn = (wave % tilesN) << 5;
  int half = lane >> 4;
  int l16  = lane & 15;

  int m = tm + l16;
  int b = m / (H * W);
  int rem = m - b * (H * W);
  int y = rem / W;
  int x = rem - y * W;

  const uint8_t* bw0 = (const uint8_t*)wgt + (size_t)(tn + l16) * K + (half << 4);
  const uint8_t* bw1 = bw0 + (size_t)16 * K;

  v8i c0 = {}, c1 = {};
  for (int r = 0; r < 9; ++r) {
    const uint8_t* arow = (const uint8_t*)act +
        (size_t)((b * Hp + y + r / 3) * Wp + x + r % 3) * Ci + (half << 4);
    const uint8_t* br0 = bw0 + r * Ci;
    const uint8_t* br1 = bw1 + r * Ci;
    for (int ci0 = 0; ci0 < Ci; ci0 += 128) {
      int2 pa01 = *(const int2*)(arow + ci0);
      int2 pa23 = *(const int2*)(arow + ci0 + 8);
      int2 pa45 = *(const int2*)(arow + ci0 + 32);
      int2 pa67 = *(const int2*)(arow + ci0 + 40);
      int4 pb0l = *(const int4*)(br0 + ci0);
      int4 pb0h = *(const int4*)(br0 + ci0 + 32);
      int4 pb1l = *(const int4*)(br1 + ci0);
      int4 pb1h = *(const int4*)(br1 + ci0 + 32);
      int2 qa01 = *(const int2*)(arow + ci0 + 64);
      int2 qa23 = *(const int2*)(arow + ci0 + 72);
      int2 qa45 = *(const int2*)(arow + ci0 + 96);
      int2 qa67 = *(const int2*)(arow + ci0 + 104);
      int4 qb0l = *(const int4*)(br0 + ci0 + 64);
      int4 qb0h = *(const int4*)(br0 + ci0 + 96);
      int4 qb1l = *(const int4*)(br1 + ci0 + 64);
      int4 qb1h = *(const int4*)(br1 + ci0 + 96);

      v8i a = packAu(pa01, pa23, pa45, pa67);
      c0 = __builtin_amdgcn_wmma_i32_16x16x64_iu8(true, a, true,
            packB(pb0l, pb0h), c0, false, false);
      c1 = __builtin_amdgcn_wmma_i32_16x16x64_iu8(true, a, true,
            packB(pb1l, pb1h), c1, false, false);
      v8i aq = packAu(qa01, qa23, qa45, qa67);
      c0 = __builtin_amdgcn_wmma_i32_16x16x64_iu8(true, aq, true,
            packB(qb0l, qb0h), c0, false, false);
      c1 = __builtin_amdgcn_wmma_i32_16x16x64_iu8(true, aq, true,
            packB(qb1l, qb1h), c1, false, false);
    }
  }
  int* o = acc + (size_t)tm * Co + tn;
#pragma unroll
  for (int r8 = 0; r8 < 8; ++r8) {
    size_t row = (size_t)(r8 + half * 8) * Co;
    o[row + l16]      = c0[r8];
    o[row + 16 + l16] = c1[r8];
  }
}

__global__ void gemm_iu8(const int8_t* __restrict__ A,
                         const int8_t* __restrict__ Bw,
                         int* __restrict__ acc, int M, int N, int K) {
  const int tilesN = N >> 5;
  const int nTiles = (M >> 4) * tilesN;
  int wave = (int)((blockIdx.x * blockDim.x + threadIdx.x) >> 5);
  int lane = threadIdx.x & 31;
  if (wave >= nTiles) return;
  int tm = (wave / tilesN) << 4;
  int tn = (wave % tilesN) << 5;
  int half = lane >> 4;
  int l16  = lane & 15;
  const uint8_t* arow = (const uint8_t*)A + (size_t)(tm + l16) * K + (half << 4);
  const uint8_t* br0  = (const uint8_t*)Bw + (size_t)(tn + l16) * K + (half << 4);
  const uint8_t* br1  = br0 + (size_t)16 * K;

  v8i c0 = {}, c1 = {};
  for (int k0 = 0; k0 < K; k0 += 128) {
    int2 pa01 = *(const int2*)(arow + k0);
    int2 pa23 = *(const int2*)(arow + k0 + 8);
    int2 pa45 = *(const int2*)(arow + k0 + 32);
    int2 pa67 = *(const int2*)(arow + k0 + 40);
    int4 pb0l = *(const int4*)(br0 + k0);
    int4 pb0h = *(const int4*)(br0 + k0 + 32);
    int4 pb1l = *(const int4*)(br1 + k0);
    int4 pb1h = *(const int4*)(br1 + k0 + 32);
    int2 qa01 = *(const int2*)(arow + k0 + 64);
    int2 qa23 = *(const int2*)(arow + k0 + 72);
    int2 qa45 = *(const int2*)(arow + k0 + 96);
    int2 qa67 = *(const int2*)(arow + k0 + 104);
    int4 qb0l = *(const int4*)(br0 + k0 + 64);
    int4 qb0h = *(const int4*)(br0 + k0 + 96);
    int4 qb1l = *(const int4*)(br1 + k0 + 64);
    int4 qb1h = *(const int4*)(br1 + k0 + 96);

    v8i a = packAu(pa01, pa23, pa45, pa67);
    c0 = __builtin_amdgcn_wmma_i32_16x16x64_iu8(true, a, true,
          packB(pb0l, pb0h), c0, false, false);
    c1 = __builtin_amdgcn_wmma_i32_16x16x64_iu8(true, a, true,
          packB(pb1l, pb1h), c1, false, false);
    v8i aq = packAu(qa01, qa23, qa45, qa67);
    c0 = __builtin_amdgcn_wmma_i32_16x16x64_iu8(true, aq, true,
          packB(qb0l, qb0h), c0, false, false);
    c1 = __builtin_amdgcn_wmma_i32_16x16x64_iu8(true, aq, true,
          packB(qb1l, qb1h), c1, false, false);
  }
  int* o = acc + (size_t)tm * N + tn;
#pragma unroll
  for (int r8 = 0; r8 < 8; ++r8) {
    size_t row = (size_t)(r8 + half * 8) * N;
    o[row + l16]      = c0[r8];
    o[row + 16 + l16] = c1[r8];
  }
}

__global__ void conv0_direct(const float* __restrict__ xin,
                             const float* __restrict__ w,
                             const float* __restrict__ bias,
                             float* __restrict__ yout) {
  const int C = 384, H = 32, W = 32;
  int idx = blockIdx.x * blockDim.x + threadIdx.x;
  int total = 64 * H * W * C;
  if (idx >= total) return;
  int co = idx % C;
  int t = idx / C;
  int x = t % W; t /= W;
  int y = t % H; t /= H;
  int b = t;
  float s = bias[co];
#pragma unroll
  for (int ci = 0; ci < 3; ++ci) {
#pragma unroll
    for (int ky = 0; ky < 3; ++ky) {
      int yy = y + ky - 1;
      if (yy < 0 || yy >= H) continue;
#pragma unroll
      for (int kx = 0; kx < 3; ++kx) {
        int xx = x + kx - 1;
        if (xx < 0 || xx >= W) continue;
        s += xin[((b * 3 + ci) * H + yy) * W + xx] *
             w[((co * 3 + ci) * 3 + ky) * 3 + kx];
      }
    }
  }
  yout[idx] = s;
}

__global__ void stats_channel(const float* __restrict__ y, int Mrows, int C,
                              float* __restrict__ mean,
                              float* __restrict__ rstd) {
  int c = blockIdx.x;
  __shared__ float ss[256], ss2[256];
  float s = 0.f, s2 = 0.f;
  for (int i = threadIdx.x; i < Mrows; i += 256) {
    float v = y[(size_t)i * C + c];
    s += v; s2 += v * v;
  }
  ss[threadIdx.x] = s; ss2[threadIdx.x] = s2;
  __syncthreads();
  for (int off = 128; off > 0; off >>= 1) {
    if ((int)threadIdx.x < off) {
      ss[threadIdx.x]  += ss[threadIdx.x + off];
      ss2[threadIdx.x] += ss2[threadIdx.x + off];
    }
    __syncthreads();
  }
  if (threadIdx.x == 0) {
    float m = ss[0] / (float)Mrows;
    float var = ss2[0] / (float)Mrows - m * m;
    mean[c] = m;
    rstd[c] = rsqrtf(var + 1e-5f);
  }
}

__global__ void pool2_bias(const int* __restrict__ acc,
                           const float* __restrict__ bias,
                           float* __restrict__ yout,
                           int Bn, int H, int W, int C) {
  int Ho = H >> 1, Wo = W >> 1;
  size_t total = (size_t)Bn * Ho * Wo * C;
  size_t idx = (size_t)blockIdx.x * blockDim.x + threadIdx.x;
  if (idx >= total) return;
  int c = (int)(idx % C); size_t t = idx / C;
  int xo = (int)(t % Wo); t /= Wo;
  int yo = (int)(t % Ho); t /= Ho;
  int b = (int)t;
  const int* p = acc + (((size_t)(b * H + 2 * yo) * W + 2 * xo) * C) + c;
  int m0 = p[0], m1 = p[C];
  int m2 = p[(size_t)W * C], m3 = p[(size_t)W * C + C];
  int mx = m0 > m1 ? m0 : m1;
  int my = m2 > m3 ? m2 : m3;
  int mm = mx > my ? mx : my;
  yout[idx] = (float)mm + bias[c];
}

__global__ void acc_bias_f32(const int* __restrict__ acc,
                             const float* __restrict__ bias,
                             float* __restrict__ yout, size_t total, int C) {
  size_t idx = (size_t)blockIdx.x * blockDim.x + threadIdx.x;
  if (idx >= total) return;
  int c = (int)(idx % C);
  float b = bias ? bias[c] : 0.f;
  yout[idx] = (float)acc[idx] + b;
}

__global__ void bn_ht_sign_pad(const float* __restrict__ y,
                               const float* __restrict__ g,
                               const float* __restrict__ bta,
                               const float* __restrict__ mean,
                               const float* __restrict__ rstd,
                               int8_t* __restrict__ out,
                               int Bn, int H, int W, int C) {
  size_t total = (size_t)Bn * H * W * C;
  size_t idx = (size_t)blockIdx.x * blockDim.x + threadIdx.x;
  if (idx >= total) return;
  int c = (int)(idx % C); size_t t = idx / C;
  int x = (int)(t % W); t /= W;
  int yy = (int)(t % H); t /= H;
  int b = (int)t;
  float z = g[c] * ((y[idx] - mean[c]) * rstd[c]) + bta[c];
  int Hp = H + 2, Wp = W + 2;
  out[((size_t)(b * Hp + yy + 1) * Wp + x + 1) * C + c] = sgn8(z);
}

__global__ void bn_ht_sign(const float* __restrict__ y,
                           const float* __restrict__ g,
                           const float* __restrict__ bta,
                           const float* __restrict__ mean,
                           const float* __restrict__ rstd,
                           int8_t* __restrict__ out, size_t total, int C) {
  size_t idx = (size_t)blockIdx.x * blockDim.x + threadIdx.x;
  if (idx >= total) return;
  int c = (int)(idx % C);
  float z = g[c] * ((y[idx] - mean[c]) * rstd[c]) + bta[c];
  out[idx] = sgn8(z);
}

__global__ void bn_ht_f32(const float* __restrict__ y,
                          const float* __restrict__ g,
                          const float* __restrict__ bta,
                          const float* __restrict__ mean,
                          const float* __restrict__ rstd,
                          float* __restrict__ out, size_t total, int C) {
  size_t idx = (size_t)blockIdx.x * blockDim.x + threadIdx.x;
  if (idx >= total) return;
  int c = (int)(idx % C);
  float z = g[c] * ((y[idx] - mean[c]) * rstd[c]) + bta[c];
  out[idx] = fminf(1.f, fmaxf(-1.f, z));
}

__global__ void bin_conv_w(const float* __restrict__ w, int8_t* __restrict__ o,
                           int Co, int Ci) {
  size_t total = (size_t)Co * 9 * Ci;
  size_t idx = (size_t)blockIdx.x * blockDim.x + threadIdx.x;
  if (idx >= total) return;
  int ci = (int)(idx % Ci); size_t t = idx / Ci;
  int r = (int)(t % 9); int co = (int)(t / 9);
  o[idx] = sgn8(w[((size_t)co * Ci + ci) * 9 + r]);
}

__global__ void bin_fc0_w(const float* __restrict__ w, int8_t* __restrict__ o) {
  size_t total = (size_t)1024 * 8192;
  size_t idx = (size_t)blockIdx.x * blockDim.x + threadIdx.x;
  if (idx >= total) return;
  int k = (int)(idx % 8192); int n = (int)(idx / 8192);
  int yx = k / 512; int c = k % 512;
  o[idx] = sgn8(w[(size_t)n * 8192 + c * 16 + yx]);
}

__global__ void bin_plain(const float* __restrict__ w, int8_t* __restrict__ o,
                          size_t total) {
  size_t idx = (size_t)blockIdx.x * blockDim.x + threadIdx.x;
  if (idx >= total) return;
  o[idx] = sgn8(w[idx]);
}

__global__ void fc2_gemm(const float* __restrict__ h,
                         const float* __restrict__ w,
                         const float* __restrict__ bias,
                         float* __restrict__ z) {
  int idx = blockIdx.x * blockDim.x + threadIdx.x;
  if (idx >= 640) return;
  int j = idx % 10, b = idx / 10;
  float s = bias[j];
  const float* hr = h + (size_t)b * 1024;
  const float* wr = w + (size_t)j * 1024;
  for (int k = 0; k < 1024; ++k) s += hr[k] * wr[k];
  z[idx] = s;
}

__global__ void bn_logsoftmax(const float* __restrict__ z,
                              float* __restrict__ out) {
  __shared__ float mean[10], rstd[10];
  int t = threadIdx.x;
  if (t < 10) {
    float s = 0.f, s2 = 0.f;
    for (int b = 0; b < 64; ++b) { float v = z[b * 10 + t]; s += v; s2 += v * v; }
    float m = s / 64.f;
    float var = s2 / 64.f - m * m;
    mean[t] = m; rstd[t] = rsqrtf(var + 1e-5f);
  }
  __syncthreads();
  float v[10]; float mx = -1e30f;
#pragma unroll
  for (int j = 0; j < 10; ++j) {
    v[j] = (z[t * 10 + j] - mean[j]) * rstd[j];
    mx = fmaxf(mx, v[j]);
  }
  float se = 0.f;
#pragma unroll
  for (int j = 0; j < 10; ++j) se += __expf(v[j] - mx);
  float lse = mx + logf(se);
#pragma unroll
  for (int j = 0; j < 10; ++j) out[t * 10 + j] = v[j] - lse;
}

extern "C" void kernel_launch(void* const* d_in, const int* in_sizes, int n_in,
                              void* d_out, int out_size, void* d_ws,
                              size_t ws_size, hipStream_t stream) {
  (void)in_sizes; (void)n_in; (void)out_size; (void)ws_size;
  const float* x = (const float*)d_in[0];
  const float* cw[6], *cb[6], *bg[6], *bb[6];
  for (int i = 0; i < 6; ++i) {
    cw[i] = (const float*)d_in[1 + 4 * i];
    cb[i] = (const float*)d_in[2 + 4 * i];
    bg[i] = (const float*)d_in[3 + 4 * i];
    bb[i] = (const float*)d_in[4 + 4 * i];
  }
  const float* fc0_w = (const float*)d_in[25];
  const float* bn6_g = (const float*)d_in[26];
  const float* bn6_b = (const float*)d_in[27];
  const float* fc1_w = (const float*)d_in[28];
  const float* fc1_b = (const float*)d_in[29];
  const float* bn7_g = (const float*)d_in[30];
  const float* bn7_b = (const float*)d_in[31];
  const float* fc2_w = (const float*)d_in[32];
  const float* fc2_b = (const float*)d_in[33];
  float* out = (float*)d_out;

  uint8_t* ws = (uint8_t*)d_ws;
  size_t off = 0;
  auto alloc = [&](size_t bytes) -> void* {
    void* p = ws + off;
    off += (bytes + 255) & ~(size_t)255;
    return p;
  };
  const size_t PA_BYTES = (size_t)64 * 34 * 34 * 384;
  const size_t P2_BYTES = (size_t)64 * 18 * 18 * 768;
  const size_t P4_BYTES = (size_t)64 * 10 * 10 * 1536;
  const size_t P1_BYTES = (size_t)64 * 18 * 18 * 384;
  const size_t P3_BYTES = (size_t)64 * 10 * 10 * 768;

  float* R1f = (float*)alloc((size_t)65536 * 384 * 4);
  int*   R1i = (int*)R1f;
  float* R2  = (float*)alloc((size_t)16384 * 768 * 4);
  int8_t* PA = (int8_t*)alloc(PA_BYTES);
  int8_t* PB = (int8_t*)alloc(P1_BYTES);
  int8_t* SF = (int8_t*)alloc((size_t)64 * 8192);
  int8_t* SG = (int8_t*)alloc((size_t)64 * 1024);
  int8_t* wb1 = (int8_t*)alloc((size_t)384 * 3456);
  int8_t* wb2 = (int8_t*)alloc((size_t)768 * 3456);
  int8_t* wb3 = (int8_t*)alloc((size_t)768 * 6912);
  int8_t* wb4 = (int8_t*)alloc((size_t)1536 * 6912);
  int8_t* wb5 = (int8_t*)alloc((size_t)512 * 13824);
  int8_t* wf0 = (int8_t*)alloc((size_t)1024 * 8192);
  int8_t* wf1 = (int8_t*)alloc((size_t)1024 * 1024);
  float* mean = (float*)alloc(1536 * 4);
  float* rstd = (float*)alloc(1536 * 4);
  int*   fca  = (int*)alloc((size_t)64 * 1024 * 4);
  float* fcy  = (float*)alloc((size_t)64 * 1024 * 4);
  float* h7   = (float*)alloc((size_t)64 * 1024 * 4);
  float* z2   = (float*)alloc((size_t)64 * 10 * 4);

  auto blks = [](size_t total, int bs) -> unsigned {
    return (unsigned)((total + bs - 1) / bs);
  };
  auto gemmBlks = [&](int M, int N) -> unsigned {
    size_t waves = ((size_t)M >> 4) * ((size_t)N >> 5);
    return blks(waves * 32, 256);
  };

  bin_conv_w<<<blks((size_t)384 * 9 * 384, 256), 256, 0, stream>>>(cw[1], wb1, 384, 384);
  bin_conv_w<<<blks((size_t)768 * 9 * 384, 256), 256, 0, stream>>>(cw[2], wb2, 768, 384);
  bin_conv_w<<<blks((size_t)768 * 9 * 768, 256), 256, 0, stream>>>(cw[3], wb3, 768, 768);
  bin_conv_w<<<blks((size_t)1536 * 9 * 768, 256), 256, 0, stream>>>(cw[4], wb4, 1536, 768);
  bin_conv_w<<<blks((size_t)512 * 9 * 1536, 256), 256, 0, stream>>>(cw[5], wb5, 512, 1536);
  bin_fc0_w<<<blks((size_t)1024 * 8192, 256), 256, 0, stream>>>(fc0_w, wf0);
  bin_plain<<<blks((size_t)1024 * 1024, 256), 256, 0, stream>>>(fc1_w, wf1, (size_t)1024 * 1024);

  conv0_direct<<<blks((size_t)64 * 32 * 32 * 384, 256), 256, 0, stream>>>(x, cw[0], cb[0], R1f);
  stats_channel<<<384, 256, 0, stream>>>(R1f, 65536, 384, mean, rstd);
  hipMemsetAsync(PA, 0, PA_BYTES, stream);
  bn_ht_sign_pad<<<blks((size_t)65536 * 384, 256), 256, 0, stream>>>(
      R1f, bg[0], bb[0], mean, rstd, PA, 64, 32, 32, 384);

  bconv_gemm_iu8<<<gemmBlks(65536, 384), 256, 0, stream>>>(PA, wb1, R1i, 64, 32, 32, 384, 384);
  pool2_bias<<<blks((size_t)16384 * 384, 256), 256, 0, stream>>>(R1i, cb[1], R2, 64, 32, 32, 384);
  stats_channel<<<384, 256, 0, stream>>>(R2, 16384, 384, mean, rstd);
  hipMemsetAsync(PB, 0, P1_BYTES, stream);
  bn_ht_sign_pad<<<blks((size_t)16384 * 384, 256), 256, 0, stream>>>(
      R2, bg[1], bb[1], mean, rstd, PB, 64, 16, 16, 384);

  bconv_gemm_iu8<<<gemmBlks(16384, 768), 256, 0, stream>>>(PB, wb2, R1i, 64, 16, 16, 384, 768);
  acc_bias_f32<<<blks((size_t)16384 * 768, 256), 256, 0, stream>>>(R1i, cb[2], R2, (size_t)16384 * 768, 768);
  stats_channel<<<768, 256, 0, stream>>>(R2, 16384, 768, mean, rstd);
  hipMemsetAsync(PA, 0, P2_BYTES, stream);
  bn_ht_sign_pad<<<blks((size_t)16384 * 768, 256), 256, 0, stream>>>(
      R2, bg[2], bb[2], mean, rstd, PA, 64, 16, 16, 768);

  bconv_gemm_iu8<<<gemmBlks(16384, 768), 256, 0, stream>>>(PA, wb3, R1i, 64, 16, 16, 768, 768);
  pool2_bias<<<blks((size_t)4096 * 768, 256), 256, 0, stream>>>(R1i, cb[3], R2, 64, 16, 16, 768);
  stats_channel<<<768, 256, 0, stream>>>(R2, 4096, 768, mean, rstd);
  hipMemsetAsync(PB, 0, P3_BYTES, stream);
  bn_ht_sign_pad<<<blks((size_t)4096 * 768, 256), 256, 0, stream>>>(
      R2, bg[3], bb[3], mean, rstd, PB, 64, 8, 8, 768);

  bconv_gemm_iu8<<<gemmBlks(4096, 1536), 256, 0, stream>>>(PB, wb4, R1i, 64, 8, 8, 768, 1536);
  acc_bias_f32<<<blks((size_t)4096 * 1536, 256), 256, 0, stream>>>(R1i, cb[4], R2, (size_t)4096 * 1536, 1536);
  stats_channel<<<1536, 256, 0, stream>>>(R2, 4096, 1536, mean, rstd);
  hipMemsetAsync(PA, 0, P4_BYTES, stream);
  bn_ht_sign_pad<<<blks((size_t)4096 * 1536, 256), 256, 0, stream>>>(
      R2, bg[4], bb[4], mean, rstd, PA, 64, 8, 8, 1536);

  bconv_gemm_iu8<<<gemmBlks(4096, 512), 256, 0, stream>>>(PA, wb5, R1i, 64, 8, 8, 1536, 512);
  pool2_bias<<<blks((size_t)1024 * 512, 256), 256, 0, stream>>>(R1i, cb[5], R2, 64, 8, 8, 512);
  stats_channel<<<512, 256, 0, stream>>>(R2, 1024, 512, mean, rstd);
  bn_ht_sign<<<blks((size_t)1024 * 512, 256), 256, 0, stream>>>(
      R2, bg[5], bb[5], mean, rstd, SF, (size_t)1024 * 512, 512);

  gemm_iu8<<<gemmBlks(64, 1024), 256, 0, stream>>>(SF, wf0, fca, 64, 1024, 8192);
  acc_bias_f32<<<blks((size_t)64 * 1024, 256), 256, 0, stream>>>(fca, nullptr, fcy, (size_t)64 * 1024, 1024);
  stats_channel<<<1024, 256, 0, stream>>>(fcy, 64, 1024, mean, rstd);
  bn_ht_sign<<<blks((size_t)64 * 1024, 256), 256, 0, stream>>>(
      fcy, bn6_g, bn6_b, mean, rstd, SG, (size_t)64 * 1024, 1024);

  gemm_iu8<<<gemmBlks(64, 1024), 256, 0, stream>>>(SG, wf1, fca, 64, 1024, 1024);
  acc_bias_f32<<<blks((size_t)64 * 1024, 256), 256, 0, stream>>>(fca, fc1_b, fcy, (size_t)64 * 1024, 1024);
  stats_channel<<<1024, 256, 0, stream>>>(fcy, 64, 1024, mean, rstd);
  bn_ht_f32<<<blks((size_t)64 * 1024, 256), 256, 0, stream>>>(
      fcy, bn7_g, bn7_b, mean, rstd, h7, (size_t)64 * 1024, 1024);

  fc2_gemm<<<3, 256, 0, stream>>>(h7, fc2_w, fc2_b, z2);
  bn_logsoftmax<<<1, 64, 0, stream>>>(z2, out);
}
